// InternLM2VEDecoderLayer_31782757990667
// MI455X (gfx1250) — matrix operand order repaired and checked by reading, not compiled, not run on hardware
//
#include <hip/hip_runtime.h>
#include <stdint.h>

// ---------------------------------------------------------------------------
// Problem constants (from reference)
// ---------------------------------------------------------------------------
#define T_DIM 2048
#define H_DIM 2048
#define NH    16
#define NKV   8
#define HD    128
#define I_DIM 8192
#define EPS   1e-5f
#define NEG_BIG (-3.0e38f)

typedef __attribute__((ext_vector_type(16))) __bf16 v16bf;
typedef __attribute__((ext_vector_type(8)))  float  v8f;
typedef unsigned short ushort_t;

union AFrag { v16bf v; uint32_t u[8]; ushort_t s[16]; };

#define WMMA_BF16(A_, B_, C_) \
  __builtin_amdgcn_wmma_f32_16x16x32_bf16(false, (A_), false, (B_), (short)0, (C_), false, false)

__device__ __forceinline__ ushort_t f2bf(float x) {
  union { float f; uint32_t u; } c; c.f = x;
  uint32_t u = c.u;
  return (ushort_t)((u + 0x7FFFu + ((u >> 16) & 1u)) >> 16);   // RNE
}
__device__ __forceinline__ float bf2f(ushort_t h) {
  union { uint32_t u; float f; } c; c.u = ((uint32_t)h) << 16;
  return c.f;
}

// CDNA5 async copy global->LDS (ASYNCcnt-tracked), inline asm per ISA 15.18.3.
__device__ __forceinline__ void async_b128(uint32_t lds_off, const void* g) {
  asm volatile("global_load_async_to_lds_b128 %0, %1, off"
               :: "v"(lds_off), "v"(g) : "memory");
}
#define WAIT_ASYNC(n) asm volatile("s_wait_asynccnt %0" :: "i"(n) : "memory")

__device__ __forceinline__ uint32_t lds_off32(const void* p) {
  // flat LDS pointers carry the DS byte offset in their low 32 bits
  return (uint32_t)(uintptr_t)p;
}

// ---------------------------------------------------------------------------
// One-shot weight convert + transpose: B[K][N] fp32 -> Bt[N][K] bf16.
// 32x32 tiles through LDS; memory-bound (~650MB total @ 23.3TB/s ~= 28us).
// ---------------------------------------------------------------------------
__global__ __launch_bounds__(256)
void convt_kernel(const float* __restrict__ B, ushort_t* __restrict__ Bt,
                  int K, int N)
{
  __shared__ ushort_t t[32][33];
  const int bk = blockIdx.x * 32;
  const int bn = blockIdx.y * 32;
  const int tx = threadIdx.x & 31;
  const int ty = threadIdx.x >> 5;          // 0..7
#pragma unroll
  for (int i = 0; i < 4; ++i) {
    int kk = ty + i * 8;
    t[tx][kk] = f2bf(B[(size_t)(bk + kk) * N + bn + tx]);
  }
  __syncthreads();
#pragma unroll
  for (int i = 0; i < 4; ++i) {
    int n = ty + i * 8;
    Bt[(size_t)(bn + n) * K + bk + tx] = t[n][tx];
  }
}

// ---------------------------------------------------------------------------
// Pure-bf16 GEMM with double-buffered async LDS staging.
//   C[M,N] = A[M,K] @ Bt[N,K]^T  (+ addsrc), A/Bt bf16, C fp32 or bf16.
// Block tile 128x128, K-step 32, 256 thr = 8 waves (2M x 4N), 8 WMMA/wave/step.
// Staging: 4 global_load_async_to_lds_b128 per thread per tile; overlap via
// s_wait_asynccnt <= 4 (previous tile) while next tile streams in.
// ---------------------------------------------------------------------------
#define BM 128
#define BN 128
#define BK 32

__global__ __launch_bounds__(256)
void gemm_async(const ushort_t* __restrict__ A, const ushort_t* __restrict__ Bt,
                void* __restrict__ Cout, const float* __restrict__ addsrc,
                int c_bf16, int M, int N, int K)
{
  __shared__ __align__(16) ushort_t As[2][BM][BK];   // 2 x 8KB
  __shared__ __align__(16) ushort_t Bs[2][BN][BK];   // 2 x 8KB

  const int tid  = threadIdx.x;
  const int lane = tid & 31;
  const int wave = tid >> 5;
  const int wm   = wave >> 2;   // 0..1
  const int wn   = wave & 3;    // 0..3
  const int bm   = blockIdx.y * BM;
  const int bn   = blockIdx.x * BN;

  v8f acc[4][2];
  const v8f vzero = {0.f,0.f,0.f,0.f,0.f,0.f,0.f,0.f};
#pragma unroll
  for (int i = 0; i < 4; ++i)
#pragma unroll
    for (int j = 0; j < 2; ++j) acc[i][j] = vzero;

  // issue async copies for K-tile kt into buffer buf
  auto issue_tile = [&](int kt, int buf) {
    const int k0 = kt * BK;
#pragma unroll
    for (int i = 0; i < 2; ++i) {
      int idx = tid + i * 256;            // 0..511 b128 transfers per operand
      int r   = idx >> 2;                 // row 0..127
      int seg = (idx & 3) * 8;            // 8 bf16 = 16B segment
      async_b128(lds_off32(&As[buf][r][seg]),
                 A  + (size_t)(bm + r) * K + k0 + seg);
      async_b128(lds_off32(&Bs[buf][r][seg]),
                 Bt + (size_t)(bn + r) * K + k0 + seg);
    }
  };

  const int nk = K / BK;
  issue_tile(0, 0);

  for (int j = 0; j < nk; ++j) {
    const int cur = j & 1;
    if (j + 1 < nk) {
      issue_tile(j + 1, cur ^ 1);
      WAIT_ASYNC(4);                      // tile j complete; j+1 in flight
    } else {
      WAIT_ASYNC(0);
    }
    __syncthreads();

    // B fragments (2): lane<16 -> K 0..15, lane>=16 -> K 16..31
    AFrag bfr[2];
#pragma unroll
    for (int nt = 0; nt < 2; ++nt) {
      int col = wn * 32 + nt * 16 + (lane & 15);
      int kb  = (lane < 16) ? 0 : 16;
      const uint32_t* src = (const uint32_t*)&Bs[cur][col][kb];
#pragma unroll
      for (int jj = 0; jj < 8; ++jj) bfr[nt].u[jj] = src[jj];
    }
    // A fragments (4): contiguous K, same order as B (lane<16 -> K 0..15)
    AFrag afr[4];
#pragma unroll
    for (int mt = 0; mt < 4; ++mt) {
      int row = wm * 64 + mt * 16 + (lane & 15);
      int kb  = (lane < 16) ? 0 : 16;
#pragma unroll
      for (int jj = 0; jj < 4; ++jj)
        afr[mt].u[jj] = *(const uint32_t*)&As[cur][row][kb + 2 * jj];
#pragma unroll
      for (int jj = 0; jj < 4; ++jj)
        afr[mt].u[4 + jj] = *(const uint32_t*)&As[cur][row][8 + kb + 2 * jj];
    }
#pragma unroll
    for (int mt = 0; mt < 4; ++mt)
#pragma unroll
      for (int nt = 0; nt < 2; ++nt)
        acc[mt][nt] = WMMA_BF16(afr[mt].v, bfr[nt].v, acc[mt][nt]);
    __syncthreads();
  }

  // ---- epilogue (uniform branches hoisted out of element loops)
  const int rhalf = (lane < 16) ? 0 : 8;
  if (c_bf16) {
    ushort_t* Cb = (ushort_t*)Cout;
#pragma unroll
    for (int mt = 0; mt < 4; ++mt)
#pragma unroll
      for (int nt = 0; nt < 2; ++nt) {
        int col = bn + wn * 32 + nt * 16 + (lane & 15);
#pragma unroll
        for (int i = 0; i < 8; ++i) {
          int row = bm + wm * 64 + mt * 16 + i + rhalf;
          Cb[(size_t)row * N + col] = f2bf(acc[mt][nt][i]);
        }
      }
  } else if (addsrc) {
    float* Cf = (float*)Cout;
#pragma unroll
    for (int mt = 0; mt < 4; ++mt)
#pragma unroll
      for (int nt = 0; nt < 2; ++nt) {
        int col = bn + wn * 32 + nt * 16 + (lane & 15);
#pragma unroll
        for (int i = 0; i < 8; ++i) {
          int row = bm + wm * 64 + mt * 16 + i + rhalf;
          size_t idx = (size_t)row * N + col;
          Cf[idx] = acc[mt][nt][i] + addsrc[idx];
        }
      }
  } else {
    float* Cf = (float*)Cout;
#pragma unroll
    for (int mt = 0; mt < 4; ++mt)
#pragma unroll
      for (int nt = 0; nt < 2; ++nt) {
        int col = bn + wn * 32 + nt * 16 + (lane & 15);
#pragma unroll
        for (int i = 0; i < 8; ++i) {
          int row = bm + wm * 64 + mt * 16 + i + rhalf;
          Cf[(size_t)row * N + col] = acc[mt][nt][i];
        }
      }
  }
}

// ---------------------------------------------------------------------------
// Flash attention (causal, GQA rep=2). Grid (T/64, NH), 128 threads = 4 waves.
// Output written directly as bf16 (feeds the wo GEMM).
// ---------------------------------------------------------------------------
__global__ __launch_bounds__(128)
void attn_flash(const float* __restrict__ q, const float* __restrict__ k,
                const float* __restrict__ v, ushort_t* __restrict__ o)
{
  __shared__ ushort_t Ks[32][128];    // [key][hd]  8KB
  __shared__ ushort_t Vt[128][32];    // [hd][key]  8KB
  __shared__ ushort_t Ps[4][16][32];  // per-wave P 4KB

  const int tid  = threadIdx.x;
  const int lane = tid & 31;
  const int w    = tid >> 5;
  const int head = blockIdx.y;
  const int kvh  = head >> 1;            // NH/NKV = 2
  const int qbase = blockIdx.x * 64;
  const int qrow  = qbase + w * 16;
  const int qstride = NH * HD;           // 2048
  const int kstride = NKV * HD;          // 1024
  const float scale = 0.08838834764831845f;   // 1/sqrt(128)
  const int rhalf = (lane < 16) ? 0 : 8;

  // preload Q as 4 A-fragments (K = HD split into 4x32)
  AFrag qf[4];
  {
    const int m  = lane & 15;
    const int kb = (lane < 16) ? 0 : 16;
    const float* qp = q + (size_t)(qrow + m) * qstride + head * HD;
#pragma unroll
    for (int c = 0; c < 4; ++c) {
#pragma unroll
      for (int j = 0; j < 4; ++j) {
        float2 t0 = *(const float2*)(qp + c * 32 + kb + 2 * j);
        qf[c].s[2 * j + 0] = f2bf(t0.x);
        qf[c].s[2 * j + 1] = f2bf(t0.y);
        float2 t1 = *(const float2*)(qp + c * 32 + 8 + kb + 2 * j);
        qf[c].s[8 + 2 * j + 0] = f2bf(t1.x);
        qf[c].s[8 + 2 * j + 1] = f2bf(t1.y);
      }
    }
  }

  v8f oacc[8];
  const v8f vzero = {0.f,0.f,0.f,0.f,0.f,0.f,0.f,0.f};
#pragma unroll
  for (int i = 0; i < 8; ++i) oacc[i] = vzero;
  float mv[8], lv[8];
#pragma unroll
  for (int i = 0; i < 8; ++i) { mv[i] = NEG_BIG; lv[i] = 0.f; }

  const int nkt = qbase / 32 + 2;        // causal bound
  for (int jt = 0; jt < nkt; ++jt) {
    const int kb0 = jt * 32;
#pragma unroll
    for (int i = 0; i < 8; ++i) {
      int f  = tid + i * 128;
      int kr = f >> 5;
      int c4 = (f & 31) * 4;
      size_t base = (size_t)(kb0 + kr) * kstride + kvh * HD + c4;
      float4 kv4 = *(const float4*)(k + base);
      Ks[kr][c4 + 0] = f2bf(kv4.x);
      Ks[kr][c4 + 1] = f2bf(kv4.y);
      Ks[kr][c4 + 2] = f2bf(kv4.z);
      Ks[kr][c4 + 3] = f2bf(kv4.w);
      float4 vv4 = *(const float4*)(v + base);
      Vt[c4 + 0][kr] = f2bf(vv4.x);
      Vt[c4 + 1][kr] = f2bf(vv4.y);
      Vt[c4 + 2][kr] = f2bf(vv4.z);
      Vt[c4 + 3][kr] = f2bf(vv4.w);
    }
    __syncthreads();

    v8f sc[2];
#pragma unroll
    for (int nt = 0; nt < 2; ++nt) {
      v8f sacc = vzero;
      int key = nt * 16 + (lane & 15);
      int kb  = (lane < 16) ? 0 : 16;
#pragma unroll
      for (int c = 0; c < 4; ++c) {
        AFrag bfr;
        const uint32_t* p = (const uint32_t*)&Ks[key][c * 32 + kb];
#pragma unroll
        for (int j = 0; j < 8; ++j) bfr.u[j] = p[j];
        sacc = WMMA_BF16(qf[c].v, bfr.v, sacc);
      }
      sc[nt] = sacc;
    }

#pragma unroll
    for (int nt = 0; nt < 2; ++nt) {
      int keyi = kb0 + nt * 16 + (lane & 15);
#pragma unroll
      for (int i = 0; i < 8; ++i) {
        int rowi = qrow + i + rhalf;
        float sv = sc[nt][i] * scale;
        sc[nt][i] = (keyi <= rowi) ? sv : NEG_BIG;
      }
    }

#pragma unroll
    for (int i = 0; i < 8; ++i) {
      float r = fmaxf(sc[0][i], sc[1][i]);
      r = fmaxf(r, __shfl_xor(r, 1, 32));
      r = fmaxf(r, __shfl_xor(r, 2, 32));
      r = fmaxf(r, __shfl_xor(r, 4, 32));
      r = fmaxf(r, __shfl_xor(r, 8, 32));
      float mn = fmaxf(mv[i], r);
      float alpha = __expf(mv[i] - mn);
      mv[i] = mn;
      float p0 = __expf(sc[0][i] - mn);
      float p1 = __expf(sc[1][i] - mn);
      sc[0][i] = p0; sc[1][i] = p1;
      float rs = p0 + p1;
      rs += __shfl_xor(rs, 1, 32);
      rs += __shfl_xor(rs, 2, 32);
      rs += __shfl_xor(rs, 4, 32);
      rs += __shfl_xor(rs, 8, 32);
      lv[i] = lv[i] * alpha + rs;
#pragma unroll
      for (int h8 = 0; h8 < 8; ++h8) oacc[h8][i] *= alpha;
    }

#pragma unroll
    for (int nt = 0; nt < 2; ++nt)
#pragma unroll
      for (int i = 0; i < 8; ++i)
        Ps[w][i + rhalf][nt * 16 + (lane & 15)] = f2bf(sc[nt][i]);

    AFrag pf;
    {
      int rowl = lane & 15;
      int kbb  = (lane < 16) ? 0 : 16;
#pragma unroll
      for (int j = 0; j < 4; ++j)
        pf.u[j] = *(const uint32_t*)&Ps[w][rowl][kbb + 2 * j];
#pragma unroll
      for (int j = 0; j < 4; ++j)
        pf.u[4 + j] = *(const uint32_t*)&Ps[w][rowl][8 + kbb + 2 * j];
    }

#pragma unroll
    for (int h8 = 0; h8 < 8; ++h8) {
      AFrag vf;
      int n  = h8 * 16 + (lane & 15);
      int kb = (lane < 16) ? 0 : 16;
      const uint32_t* p = (const uint32_t*)&Vt[n][kb];
#pragma unroll
      for (int j = 0; j < 8; ++j) vf.u[j] = p[j];
      oacc[h8] = WMMA_BF16(pf.v, vf.v, oacc[h8]);
    }
    __syncthreads();
  }

#pragma unroll
  for (int h8 = 0; h8 < 8; ++h8) {
    int col = h8 * 16 + (lane & 15);
#pragma unroll
    for (int i = 0; i < 8; ++i) {
      int row = qrow + i + rhalf;
      o[(size_t)row * qstride + head * HD + col] = f2bf(oacc[h8][i] / lv[i]);
    }
  }
}

// ---------------------------------------------------------------------------
// Fused (optional add) + RMSNorm; norm output emitted as bf16 (GEMM A operand)
// ---------------------------------------------------------------------------
__global__ __launch_bounds__(256)
void add_rmsnorm(const float* __restrict__ a, const float* __restrict__ b,
                 const float* __restrict__ w, float* __restrict__ resid_out,
                 ushort_t* __restrict__ norm_out)
{
  __shared__ float sred[8];
  const int row  = blockIdx.x;
  const int tid  = threadIdx.x;
  const int lane = tid & 31;
  const int wv   = tid >> 5;
  const size_t base = (size_t)row * H_DIM;

  float x[8];
#pragma unroll
  for (int i = 0; i < 8; ++i) x[i] = a[base + tid + i * 256];
  if (b) {
#pragma unroll
    for (int i = 0; i < 8; ++i) x[i] += b[base + tid + i * 256];
  }
  if (resid_out) {
#pragma unroll
    for (int i = 0; i < 8; ++i) resid_out[base + tid + i * 256] = x[i];
  }
  float ss = 0.f;
#pragma unroll
  for (int i = 0; i < 8; ++i) ss += x[i] * x[i];
  for (int m = 1; m < 32; m <<= 1) ss += __shfl_xor(ss, m, 32);
  if (lane == 0) sred[wv] = ss;
  __syncthreads();
  float tot = 0.f;
#pragma unroll
  for (int i = 0; i < 8; ++i) tot += sred[i];
  float inv = rsqrtf(tot / (float)H_DIM + EPS);
#pragma unroll
  for (int i = 0; i < 8; ++i)
    norm_out[base + tid + i * 256] = f2bf(x[i] * inv * w[tid + i * 256]);
}

// ---------------------------------------------------------------------------
// RoPE in-place (fp32 q/k). Grid (T, nheads), 64 threads.
// ---------------------------------------------------------------------------
__global__ __launch_bounds__(64)
void rope_kernel(float* __restrict__ x, const int* __restrict__ pos, int nheads)
{
  const int t = blockIdx.x, h = blockIdx.y, i = threadIdx.x;
  float inv = __powf(10000.0f, -(float)i / 64.0f);
  float ang = (float)pos[t] * inv;
  float s, c;
  __sincosf(ang, &s, &c);
  float* p = x + (size_t)t * nheads * HD + h * HD;
  float x1 = p[i], x2 = p[i + 64];
  p[i]      = x1 * c - x2 * s;
  p[i + 64] = x2 * c + x1 * s;
}

// ---------------------------------------------------------------------------
// SwiGLU gate on bf16: h1 = silu(h1) * h3 (in place)
// ---------------------------------------------------------------------------
__global__ __launch_bounds__(256)
void silu_mul_bf(ushort_t* __restrict__ h1, const ushort_t* __restrict__ h3, int n)
{
  int i = blockIdx.x * 256 + threadIdx.x;
  if (i < n) {
    float a = bf2f(h1[i]);
    float g = a / (1.f + __expf(-a));
    h1[i] = f2bf(g * bf2f(h3[i]));
  }
}

// ---------------------------------------------------------------------------
// Per-token expert select: out = mask[t] ? vis : txt
// ---------------------------------------------------------------------------
__global__ __launch_bounds__(256)
void select_kernel(const unsigned char* __restrict__ mask,
                   const float* __restrict__ vis, const float* __restrict__ txt,
                   float* __restrict__ out, int n)
{
  int i = blockIdx.x * 256 + threadIdx.x;
  if (i < n) {
    int t = i / H_DIM;
    out[i] = mask[t] ? vis[i] : txt[i];
  }
}

// ---------------------------------------------------------------------------
// Host orchestration
// ---------------------------------------------------------------------------
extern "C" void kernel_launch(void* const* d_in, const int* in_sizes, int n_in,
                              void* d_out, int out_size, void* d_ws, size_t ws_size,
                              hipStream_t stream)
{
  (void)in_sizes; (void)n_in; (void)out_size; (void)ws_size;

  const int*   positions = (const int*)d_in[0];
  const float* hidden    = (const float*)d_in[1];
  const float* residual  = (const float*)d_in[2];
  const unsigned char* vmask = (const unsigned char*)d_in[3];
  const float* attn_nw = (const float*)d_in[4];
  const float* ffn_nw  = (const float*)d_in[5];
  const float* wq  = (const float*)d_in[6];
  const float* wk  = (const float*)d_in[7];
  const float* wv  = (const float*)d_in[8];
  const float* wo  = (const float*)d_in[9];
  const float* w1  = (const float*)d_in[10];
  const float* w3  = (const float*)d_in[11];
  const float* w2  = (const float*)d_in[12];
  const float* w1v = (const float*)d_in[13];
  const float* w3v = (const float*)d_in[14];
  const float* w2v = (const float*)d_in[15];

  float* out_hs    = (float*)d_out;
  float* out_resid = (float*)d_out + (size_t)T_DIM * H_DIM;

  const size_t TH = (size_t)T_DIM * H_DIM;   // 4M elems
  const size_t TI = (size_t)T_DIM * I_DIM;   // 16M elems
  const size_t TKV = (size_t)T_DIM * NKV * HD;

  // ---- workspace carve-out (bytes)
  unsigned char* p = (unsigned char*)d_ws;
  auto carve = [&](size_t bytes) { void* r = p; p += (bytes + 255) & ~(size_t)255; return r; };

  ushort_t* wq_t  = (ushort_t*)carve((size_t)H_DIM * NH * HD * 2);
  ushort_t* wk_t  = (ushort_t*)carve((size_t)H_DIM * NKV * HD * 2);
  ushort_t* wv_t  = (ushort_t*)carve((size_t)H_DIM * NKV * HD * 2);
  ushort_t* wo_t  = (ushort_t*)carve((size_t)NH * HD * H_DIM * 2);
  ushort_t* w1_t  = (ushort_t*)carve((size_t)H_DIM * I_DIM * 2);
  ushort_t* w3_t  = (ushort_t*)carve((size_t)H_DIM * I_DIM * 2);
  ushort_t* w2_t  = (ushort_t*)carve((size_t)I_DIM * H_DIM * 2);
  ushort_t* w1v_t = (ushort_t*)carve((size_t)H_DIM * I_DIM * 2);
  ushort_t* w3v_t = (ushort_t*)carve((size_t)H_DIM * I_DIM * 2);
  ushort_t* w2v_t = (ushort_t*)carve((size_t)I_DIM * H_DIM * 2);
  ushort_t* xnorm_bf = (ushort_t*)carve(TH * 2);
  ushort_t* attn_bf  = (ushort_t*)carve(TH * 2);
  ushort_t* h1_bf    = (ushort_t*)carve(TI * 2);
  ushort_t* h3_bf    = (ushort_t*)carve(TI * 2);
  float* resid1 = (float*)carve(TH * 4);
  float* qbuf   = (float*)carve(TH * 4);
  float* kbuf   = (float*)carve(TKV * 4);
  float* vbuf   = (float*)carve(TKV * 4);
  float* ytxt   = (float*)carve(TH * 4);
  float* yvis   = (float*)carve(TH * 4);

  // ---- 0) weight convert + transpose (bf16 [N][K])
  convt_kernel<<<dim3(H_DIM/32, (NH*HD)/32),  256, 0, stream>>>(wq,  wq_t,  H_DIM, NH*HD);
  convt_kernel<<<dim3(H_DIM/32, (NKV*HD)/32), 256, 0, stream>>>(wk,  wk_t,  H_DIM, NKV*HD);
  convt_kernel<<<dim3(H_DIM/32, (NKV*HD)/32), 256, 0, stream>>>(wv,  wv_t,  H_DIM, NKV*HD);
  convt_kernel<<<dim3((NH*HD)/32, H_DIM/32),  256, 0, stream>>>(wo,  wo_t,  NH*HD, H_DIM);
  convt_kernel<<<dim3(H_DIM/32, I_DIM/32),    256, 0, stream>>>(w1,  w1_t,  H_DIM, I_DIM);
  convt_kernel<<<dim3(H_DIM/32, I_DIM/32),    256, 0, stream>>>(w3,  w3_t,  H_DIM, I_DIM);
  convt_kernel<<<dim3(I_DIM/32, H_DIM/32),    256, 0, stream>>>(w2,  w2_t,  I_DIM, H_DIM);
  convt_kernel<<<dim3(H_DIM/32, I_DIM/32),    256, 0, stream>>>(w1v, w1v_t, H_DIM, I_DIM);
  convt_kernel<<<dim3(H_DIM/32, I_DIM/32),    256, 0, stream>>>(w3v, w3v_t, H_DIM, I_DIM);
  convt_kernel<<<dim3(I_DIM/32, H_DIM/32),    256, 0, stream>>>(w2v, w2v_t, I_DIM, H_DIM);

  dim3 g_sq(H_DIM / BN, T_DIM / BM);
  dim3 g_kv(NKV * HD / BN, T_DIM / BM);
  dim3 g_up(I_DIM / BN, T_DIM / BM);

  // 1) resid1 = hidden + residual ; xnorm_bf = rmsnorm(resid1)*attn_norm_w
  add_rmsnorm<<<T_DIM, 256, 0, stream>>>(hidden, residual, attn_nw, resid1, xnorm_bf);

  // 2) QKV projections (fp32 out for RoPE/attention)
  gemm_async<<<g_sq, 256, 0, stream>>>(xnorm_bf, wq_t, qbuf, nullptr, 0, T_DIM, NH*HD, H_DIM);
  gemm_async<<<g_kv, 256, 0, stream>>>(xnorm_bf, wk_t, kbuf, nullptr, 0, T_DIM, NKV*HD, H_DIM);
  gemm_async<<<g_kv, 256, 0, stream>>>(xnorm_bf, wv_t, vbuf, nullptr, 0, T_DIM, NKV*HD, H_DIM);

  // 3) RoPE
  rope_kernel<<<dim3(T_DIM, NH),  64, 0, stream>>>(qbuf, positions, NH);
  rope_kernel<<<dim3(T_DIM, NKV), 64, 0, stream>>>(kbuf, positions, NKV);

  // 4) causal flash attention (bf16 out)
  attn_flash<<<dim3(T_DIM / 64, NH), 128, 0, stream>>>(qbuf, kbuf, vbuf, attn_bf);

  // 5) output projection fused with residual add -> returned residual
  gemm_async<<<g_sq, 256, 0, stream>>>(attn_bf, wo_t, out_resid, resid1, 0, T_DIM, H_DIM, NH*HD);

  // 6) second norm (bf16 out)
  add_rmsnorm<<<T_DIM, 256, 0, stream>>>(out_resid, nullptr, ffn_nw, nullptr, xnorm_bf);

  // 7) text expert MLP
  gemm_async<<<g_up, 256, 0, stream>>>(xnorm_bf, w1_t, h1_bf, nullptr, 1, T_DIM, I_DIM, H_DIM);
  gemm_async<<<g_up, 256, 0, stream>>>(xnorm_bf, w3_t, h3_bf, nullptr, 1, T_DIM, I_DIM, H_DIM);
  silu_mul_bf<<<(int)(TI / 256), 256, 0, stream>>>(h1_bf, h3_bf, (int)TI);
  gemm_async<<<g_sq, 256, 0, stream>>>(h1_bf, w2_t, ytxt, nullptr, 0, T_DIM, H_DIM, I_DIM);

  // 8) visual expert MLP
  gemm_async<<<g_up, 256, 0, stream>>>(xnorm_bf, w1v_t, h1_bf, nullptr, 1, T_DIM, I_DIM, H_DIM);
  gemm_async<<<g_up, 256, 0, stream>>>(xnorm_bf, w3v_t, h3_bf, nullptr, 1, T_DIM, I_DIM, H_DIM);
  silu_mul_bf<<<(int)(TI / 256), 256, 0, stream>>>(h1_bf, h3_bf, (int)TI);
  gemm_async<<<g_sq, 256, 0, stream>>>(h1_bf, w2v_t, yvis, nullptr, 0, T_DIM, H_DIM, I_DIM);

  // 9) per-token select
  select_kernel<<<(int)(TH / 256), 256, 0, stream>>>(vmask, yvis, ytxt, out_hs, (int)TH);
}
